// InputEmbedding_26379689132424
// MI455X (gfx1250) — compile-verified, matrix operand order checked by reading, not run on hardware
//
#include <hip/hip_runtime.h>
#include <hip/hip_fp16.h>

typedef __attribute__((ext_vector_type(16))) _Float16 v16h;
typedef __attribute__((ext_vector_type(8)))  float    v8f;

#define BB   4
#define NN   4096
#define INCH 32
#define CIN  67
#define CPAD 96
#define H0   32
#define H1   64
#define WPB  4                       // waves per block
#define NBLK ((BB * NN) / WPB)       // 4096 blocks for mlp passes

// workspace layout (bytes)
#define OFF_GI0  0u                            // B*N*16 ints = 1 MiB
#define OFF_GI1  (1u << 20)                    // B*N*32 ints = 2 MiB
#define OFF_W    (3u << 20)                    // 2*5120 halfs = 20 KiB (swizzled weights)
#define OFF_BN   ((3u << 20) + (1u << 15))     // 2*192 floats (bn scale/shift)
#define OFF_PART (4u << 20)                    // NBLK*64*2 floats = 2 MiB

// ---------------------------------------------------------------------------
// copy xyz passthrough to the first part of d_out
__global__ void copy_xyz_k(const float* __restrict__ src, float* __restrict__ dst, int n) {
    int i = blockIdx.x * blockDim.x + threadIdx.x;
    if (i < n) dst[i] = src[i];
}

// ---------------------------------------------------------------------------
// Pre-swizzle weights into the exact WMMA B-operand per-lane layout (f16).
// blockIdx.x in [0,20): scale s = bid/10; r<6 -> layer0 tile (kc=r>>1, nt=r&1),
// r>=6 -> layer1 tile nt=r-6. Each tile = 32 lanes * 16 halfs.
__global__ void prep_weights_k(const float* __restrict__ W00, const float* __restrict__ W01,
                               const float* __restrict__ W10, const float* __restrict__ W11,
                               _Float16* __restrict__ wsw) {
    int bid = blockIdx.x, lane = threadIdx.x;
    int s = bid / 10, r = bid % 10;
    const float* W0 = s ? W10 : W00;
    const float* W1 = s ? W11 : W01;
    int half = lane >> 4, col = lane & 15;
    if (r < 6) {
        int kc = r >> 1, nt = r & 1;
        _Float16* dst = wsw + s * 5120 + r * 512 + lane * 16;
        int n = nt * 16 + col;
        #pragma unroll
        for (int j = 0; j < 8; ++j) {
            int kl = ((j < 4) ? (2 * j) : (16 + 2 * (j - 4))) + 8 * half;
            #pragma unroll
            for (int e = 0; e < 2; ++e) {
                int k = kc * 32 + kl + e;
                float v = (k < CIN) ? W0[n * CIN + k] : 0.0f;   // zero-pad 67 -> 96
                dst[2 * j + e] = (_Float16)v;
            }
        }
    } else {
        int nt = r - 6;
        _Float16* dst = wsw + s * 5120 + 3072 + nt * 512 + lane * 16;
        int n = nt * 16 + col;
        #pragma unroll
        for (int j = 0; j < 8; ++j) {
            int kl = ((j < 4) ? (2 * j) : (16 + 2 * (j - 4))) + 8 * half;
            #pragma unroll
            for (int e = 0; e < 2; ++e)
                dst[2 * j + e] = (_Float16)W1[n * H0 + kl + e];
        }
    }
}

// ---------------------------------------------------------------------------
// query_ball_point: one thread per query point; ascending-index scan == sort+take-K.
__global__ void ball_query_k(const float* __restrict__ xyz, int* __restrict__ gi,
                             int K, float r2) {
    int p = blockIdx.x * blockDim.x + threadIdx.x;
    if (p >= BB * NN) return;
    int b = p >> 12, n = p & (NN - 1);
    const float* xb = xyz + (size_t)b * 3 * NN;
    float qx = xb[n], qy = xb[NN + n], qz = xb[2 * NN + n];
    int* g = gi + (size_t)p * K;
    int cnt = 0, first = n;          // self is always in-radius (d=0)
    for (int j = 0; j < NN; ++j) {
        float dx = xb[j] - qx, dy = xb[NN + j] - qy, dz = xb[2 * NN + j] - qz;
        float d = dx * dx + dy * dy + dz * dz;
        if (d <= r2) {
            if (cnt == 0) first = j;
            g[cnt++] = j;
            if (cnt == K) break;
        }
    }
    for (; cnt < K; ++cnt) g[cnt] = first;
}

// ---------------------------------------------------------------------------
// Fused grouped-feature + MLP pass. One wave per (b,n) point; K/16 m-tiles.
// stage 0: GEMM1 only, emit per-block partial sum/sumsq of y0 (deterministic).
// stage 1: GEMM1 + BN0 + ReLU + GEMM2, emit partials of y1.
// stage 2: full forward + BN1 + ReLU + max over K -> output channels.
__global__ void __launch_bounds__(WPB * 32)
mlp_pass_k(const float* __restrict__ xyz, const float* __restrict__ pts,
           const int* __restrict__ gi, int K, float inv_r,
           const _Float16* __restrict__ wts,     // this scale's swizzled weights
           const float* __restrict__ b0, const float* __restrict__ b1,
           const float* __restrict__ bn,         // 192 floats: s0|t0|s1|t1
           float* __restrict__ partial,          // NBLK * 64 * 2
           float* __restrict__ out, int chOff, int stage) {
    __shared__ _Float16 feat[WPB][16 * CPAD];
    __shared__ _Float16 zbuf[WPB][16 * H0];
    __shared__ float    sstat[WPB][64][2];

    int tid = threadIdx.x;
    int w = tid >> 5, lane = tid & 31;
    int half = lane >> 4, col = lane & 15;

    for (int i = tid; i < WPB * 64 * 2; i += WPB * 32) ((float*)sstat)[i] = 0.0f;

    int p = blockIdx.x * WPB + w;
    int b = p >> 12, n = p & (NN - 1);
    const float* pb = pts + (size_t)b * INCH * NN;
    const float* xb = xyz + (size_t)b * 3 * NN;

    float ssum[4] = {0.f, 0.f, 0.f, 0.f}, ssq[4] = {0.f, 0.f, 0.f, 0.f};
    float mx[4] = {0.f, 0.f, 0.f, 0.f};   // outputs are post-ReLU >= 0
    int ntiles = K >> 4;

    for (int t = 0; t < ntiles; ++t) {
        __syncthreads();
        // ---- build 16 x 96 f16 feature tile: lane handles sample=col, 48 channels
        {
            int sidx = gi[(size_t)p * K + t * 16 + col];
            _Float16* f = feat[w] + col * CPAD;
            int cb = half * 48;
            for (int c = cb; c < cb + 48; ++c) {
                float v;
                if (c < 32)      v = pb[c * NN + sidx] - pb[c * NN + n];        // gpts - center
                else if (c < 64) v = pb[(c - 32) * NN + n];                     // center
                else if (c < 67) v = (xb[(c - 64) * NN + sidx] - xb[(c - 64) * NN + n]) * inv_r;
                else             v = 0.0f;                                      // pad
                f[c] = (_Float16)v;
            }
        }
        __syncthreads();

        // ---- GEMM1: 16x96 * 96x32 (+bias), 3 k-chunks x 2 n-tiles of WMMA
        v8f c0[2];
        #pragma unroll
        for (int nt = 0; nt < 2; ++nt) {
            float bias = b0[nt * 16 + col];
            #pragma unroll
            for (int r = 0; r < 8; ++r) c0[nt][r] = bias;
        }
        #pragma unroll
        for (int kc = 0; kc < 3; ++kc) {
            v16h a;
            const _Float16* f = feat[w] + col * CPAD + kc * 32 + 8 * half;
            #pragma unroll
            for (int j = 0; j < 8; ++j) {
                int kl = (j < 4) ? (2 * j) : (16 + 2 * (j - 4));
                a[2 * j]     = f[kl];
                a[2 * j + 1] = f[kl + 1];
            }
            #pragma unroll
            for (int nt = 0; nt < 2; ++nt) {
                v16h bf = *(const v16h*)(wts + (kc * 2 + nt) * 512 + lane * 16);
                c0[nt] = __builtin_amdgcn_wmma_f32_16x16x32_f16(
                             false, a, false, bf, (short)0, c0[nt], false, false);
            }
        }

        if (stage == 0) {
            #pragma unroll
            for (int nt = 0; nt < 2; ++nt)
                #pragma unroll
                for (int r = 0; r < 8; ++r) { float v = c0[nt][r]; ssum[nt] += v; ssq[nt] += v * v; }
            continue;
        }

        // ---- BN0 + ReLU -> 16x32 f16 staging tile
        #pragma unroll
        for (int nt = 0; nt < 2; ++nt) {
            int ch = nt * 16 + col;
            float sc = bn[ch], sh = bn[32 + ch];
            #pragma unroll
            for (int r = 0; r < 8; ++r) {
                int m = r + 8 * half;
                float z = c0[nt][r] * sc + sh;
                zbuf[w][m * H0 + ch] = (_Float16)(z > 0.0f ? z : 0.0f);
            }
        }
        __syncthreads();

        // ---- GEMM2: 16x32 * 32x64 (+bias), 4 n-tiles of WMMA
        v16h a2;
        {
            const _Float16* f = zbuf[w] + col * H0 + 8 * half;
            #pragma unroll
            for (int j = 0; j < 8; ++j) {
                int kl = (j < 4) ? (2 * j) : (16 + 2 * (j - 4));
                a2[2 * j]     = f[kl];
                a2[2 * j + 1] = f[kl + 1];
            }
        }
        v8f c1[4];
        #pragma unroll
        for (int nt = 0; nt < 4; ++nt) {
            float bias = b1[nt * 16 + col];
            #pragma unroll
            for (int r = 0; r < 8; ++r) c1[nt][r] = bias;
            v16h bf = *(const v16h*)(wts + 3072 + nt * 512 + lane * 16);
            c1[nt] = __builtin_amdgcn_wmma_f32_16x16x32_f16(
                         false, a2, false, bf, (short)0, c1[nt], false, false);
        }

        if (stage == 1) {
            #pragma unroll
            for (int nt = 0; nt < 4; ++nt)
                #pragma unroll
                for (int r = 0; r < 8; ++r) { float v = c1[nt][r]; ssum[nt] += v; ssq[nt] += v * v; }
        } else {
            #pragma unroll
            for (int nt = 0; nt < 4; ++nt) {
                int ch = nt * 16 + col;
                float sc = bn[64 + ch], sh = bn[128 + ch];
                #pragma unroll
                for (int r = 0; r < 8; ++r) {
                    float z = c1[nt][r] * sc + sh;
                    z = z > 0.0f ? z : 0.0f;
                    mx[nt] = z > mx[nt] ? z : mx[nt];
                }
            }
        }
    }

    if (stage == 2) {
        #pragma unroll
        for (int nt = 0; nt < 4; ++nt) {
            float o = __shfl_xor(mx[nt], 16, 32);           // combine lane halves (m 0-7 | 8-15)
            mx[nt] = mx[nt] > o ? mx[nt] : o;
        }
        if (half == 0) {
            #pragma unroll
            for (int nt = 0; nt < 4; ++nt) {
                int ch = chOff + nt * 16 + col;
                out[((size_t)b * 128 + ch) * NN + n] = mx[nt];
            }
        }
    } else {
        int nts = (stage == 0) ? 2 : 4;
        for (int nt = 0; nt < nts; ++nt) {
            ssum[nt] += __shfl_xor(ssum[nt], 16, 32);
            ssq[nt]  += __shfl_xor(ssq[nt],  16, 32);
        }
        __syncthreads();
        if (half == 0)
            for (int nt = 0; nt < nts; ++nt) {
                sstat[w][nt * 16 + col][0] = ssum[nt];
                sstat[w][nt * 16 + col][1] = ssq[nt];
            }
        __syncthreads();
        if (tid < 64) {                                      // fixed-order block partial
            float s = 0.f, q = 0.f;
            for (int ww = 0; ww < WPB; ++ww) { s += sstat[ww][tid][0]; q += sstat[ww][tid][1]; }
            partial[(size_t)blockIdx.x * 128 + tid * 2]     = s;
            partial[(size_t)blockIdx.x * 128 + tid * 2 + 1] = q;
        }
    }
}

// ---------------------------------------------------------------------------
// Fixed-order reduction of block partials -> BN scale/shift (deterministic).
__global__ void finalize_bn_k(const float* __restrict__ partial, int nch, float inv_cnt,
                              const float* __restrict__ gamma, const float* __restrict__ beta,
                              float* __restrict__ bnscale, float* __restrict__ bnshift) {
    int ch = threadIdx.x;
    if (ch >= nch) return;
    float s = 0.f, q = 0.f;
    for (int blk = 0; blk < NBLK; ++blk) {
        s += partial[(size_t)blk * 128 + ch * 2];
        q += partial[(size_t)blk * 128 + ch * 2 + 1];
    }
    float mean = s * inv_cnt;
    float var  = q * inv_cnt - mean * mean;
    float sc   = gamma[ch] * rsqrtf(var + 1e-5f);
    bnscale[ch] = sc;
    bnshift[ch] = beta[ch] - mean * sc;
}

// ---------------------------------------------------------------------------
extern "C" void kernel_launch(void* const* d_in, const int* in_sizes, int n_in,
                              void* d_out, int out_size, void* d_ws, size_t ws_size,
                              hipStream_t stream) {
    const float* xyz = (const float*)d_in[0];
    const float* pts = (const float*)d_in[1];
    // params flatten (insertion order): per scale, per layer: W, b, gamma, beta
    auto P = [&](int s, int l, int which) { return (const float*)d_in[2 + s * 8 + l * 4 + which]; };

    char* ws = (char*)d_ws;
    int*      gi[2]   = { (int*)(ws + OFF_GI0), (int*)(ws + OFF_GI1) };
    _Float16* wsw     = (_Float16*)(ws + OFF_W);
    float*    bnbuf   = (float*)(ws + OFF_BN);
    float*    partial = (float*)(ws + OFF_PART);
    float*    out     = (float*)d_out;
    float*    newpts  = out + BB * 3 * NN;

    copy_xyz_k<<<(BB * 3 * NN + 255) / 256, 256, 0, stream>>>(xyz, out, BB * 3 * NN);
    prep_weights_k<<<20, 32, 0, stream>>>(P(0,0,0), P(0,1,0), P(1,0,0), P(1,1,0), wsw);

    const float radius[2] = {0.1f, 0.2f};
    const int   Ks[2]     = {16, 32};
    for (int s = 0; s < 2; ++s) {
        int K = Ks[s];
        float r = radius[s];
        ball_query_k<<<(BB * NN + 255) / 256, 256, 0, stream>>>(xyz, gi[s], K, r * r);

        const _Float16* wts = wsw + s * 5120;
        float* bn = bnbuf + s * 192;
        float inv_cnt = 1.0f / (float)((size_t)BB * NN * K);

        mlp_pass_k<<<NBLK, WPB * 32, 0, stream>>>(xyz, pts, gi[s], K, 1.0f / r, wts,
            P(s,0,1), P(s,1,1), bn, partial, newpts, s * 64, 0);
        finalize_bn_k<<<1, 64, 0, stream>>>(partial, H0, inv_cnt, P(s,0,2), P(s,0,3), bn, bn + 32);

        mlp_pass_k<<<NBLK, WPB * 32, 0, stream>>>(xyz, pts, gi[s], K, 1.0f / r, wts,
            P(s,0,1), P(s,1,1), bn, partial, newpts, s * 64, 1);
        finalize_bn_k<<<1, 64, 0, stream>>>(partial, H1, inv_cnt, P(s,1,2), P(s,1,3), bn + 64, bn + 128);

        mlp_pass_k<<<NBLK, WPB * 32, 0, stream>>>(xyz, pts, gi[s], K, 1.0f / r, wts,
            P(s,0,1), P(s,1,1), bn, partial, newpts, s * 64, 2);
    }
}
